// Transformer_18253611008558
// MI455X (gfx1250) — matrix operand order repaired and checked by reading, not compiled, not run on hardware
//
#include <hip/hip_runtime.h>
#include <hip/hip_bf16.h>
#include <math.h>

// ---------------------------------------------------------------------------
// Transformer encoder inference for MI455X (gfx1250), wave32 + WMMA f16.
// B=32 S=240 H=1024 L=8 HEADS=16 D=64 VOCAB=48
// GEMMs: 128x64 block tile, B staged via global_load_async_to_lds_b128.
// ---------------------------------------------------------------------------

typedef __attribute__((ext_vector_type(8)))  _Float16 v8h;
typedef __attribute__((ext_vector_type(16))) _Float16 v16h;
typedef __attribute__((ext_vector_type(8)))  float    v8f;

#define DI __device__ __forceinline__

constexpr int BB = 32, SS = 240, HH = 1024, LL = 8, NH = 16, DD = 64, VV = 48;
constexpr int MM = BB * SS;   // 7680 token rows

DI v8f wmma_f16(v16h a, v16h b, v8f c) {
  // D = A(16x32 f16) * B(32x16 f16) + C(16x16 f32)
  return __builtin_amdgcn_wmma_f32_16x16x32_f16(
      /*neg_a=*/false, a, /*neg_b=*/false, b,
      /*c_mod=*/(short)0, c, /*reuse_a=*/false, /*reuse_b=*/false);
}

DI v8f zero8() {
  v8f z;
#pragma unroll
  for (int i = 0; i < 8; ++i) z[i] = 0.0f;
  return z;
}

// A fragment: 16x32 f16, row-major source. p = &A[row][k0].
DI v16h loadA(const _Float16* p, int half) {
  const _Float16* q = p + half * 8;
  v8h lo = *(const v8h*)q;
  v8h hi = *(const v8h*)(q + 16);
  v16h r;
#pragma unroll
  for (int i = 0; i < 8; ++i) { r[i] = lo[i]; r[8 + i] = hi[i]; }
  return r;
}

// B fragment: 32x16 f16, memory column-major by N (Bt[n][k]). p = &Bt[col][k0].
DI v16h loadB(const _Float16* p, int half) {
  const _Float16* q = p + half * 8;
  v8h lo = *(const v8h*)q;
  v8h hi = *(const v8h*)(q + 16);
  v16h r;
#pragma unroll
  for (int i = 0; i < 8; ++i) { r[i] = lo[i]; r[8 + i] = hi[i]; }
  return r;
}

// ---------------------------------------------------------------------------
// Weight convert + transpose: W[l][k][n] f32 -> Wt[l][n][k] f16
// ---------------------------------------------------------------------------
__global__ __launch_bounds__(256) void convT_kernel(const float* __restrict__ W,
                                                    _Float16* __restrict__ Wt,
                                                    int total) {
  int idx = blockIdx.x * 256 + threadIdx.x;
  if (idx >= total) return;
  int l = idx >> 20;
  int rem = idx & ((1 << 20) - 1);
  int k = rem >> 10;
  int n = rem & 1023;
  Wt[((size_t)l << 20) + ((size_t)n << 10) + k] = (_Float16)W[idx];
}

// ---------------------------------------------------------------------------
// Embedding: argmax over VOCAB, gather, validity mask, +pos-emb
// ---------------------------------------------------------------------------
__global__ __launch_bounds__(256) void embed_kernel(const float* __restrict__ x,
                                                    const float* __restrict__ wemb,
                                                    const float* __restrict__ pemb,
                                                    const int* __restrict__ lengths,
                                                    float* __restrict__ z,
                                                    _Float16* __restrict__ zh) {
  int row = blockIdx.x;          // b*SS + s
  int b = row / SS, s = row - b * SS;
  __shared__ int tok;
  if (threadIdx.x == 0) {
    const float* xr = x + (size_t)row * VV;
    float best = xr[0];
    int bi = 0;
    for (int j = 1; j < VV; ++j) {
      float v = xr[j];
      if (v > best) { best = v; bi = j; }
    }
    tok = bi;
  }
  __syncthreads();
  float validf = (s < lengths[b]) ? 1.0f : 0.0f;
  int tk = tok;
#pragma unroll
  for (int i = 0; i < 4; ++i) {
    int h = threadIdx.x * 4 + i;
    float v = wemb[(size_t)tk * HH + h] * validf + pemb[(size_t)s * HH + h];
    z[(size_t)row * HH + h] = v;
    zh[(size_t)row * HH + h] = (_Float16)v;
  }
}

// ---------------------------------------------------------------------------
// WMMA GEMM: out[m][n] = A[m][:] dot Wt[n][:] + bias[n],  M=7680 K=N=1024
// Block = 256 threads (8 waves) computes 128 rows x 64 cols.
// B K-chunks (64x64 f16 = 8KB) staged into LDS with async-to-LDS copies.
// MODE 0: f16 out in [b][head][s][d] layout (Q / K)
// MODE 1: f16 out in [b][head][d][s] layout (V transposed)
// MODE 2: f16 out row-major with exact GELU (FFN1)
// MODE 3: f32 out row-major (FFN2)
// ---------------------------------------------------------------------------
template <int MODE>
__global__ __launch_bounds__(256) void gemm_kernel(const _Float16* __restrict__ A,
                                                   const _Float16* __restrict__ Bt,
                                                   const float* __restrict__ bias,
                                                   void* __restrict__ outp) {
  __shared__ _Float16 ldsB[64 * 64];   // [n_local][k_local], row stride 128B

  const int tid  = threadIdx.x;
  const int lane = tid & 31;
  const int wid  = tid >> 5;
  const int half = lane >> 4;
  const int lm   = lane & 15;
  const int m0 = blockIdx.x * 128 + wid * 16;   // wave's 16 rows
  const int n0 = blockIdx.y * 64;               // block's 64 cols

  const _Float16* Arow = A + (size_t)(m0 + lm) * HH;

  // Async staging plan: thread copies 32B of the 8KB chunk.
  // row = tid/4 (0..63), col chunk = (tid%4)*16 halves.
  const int srow = tid >> 2;
  const int scol = (tid & 3) * 16;
  const unsigned ldsaddr = (unsigned)(uintptr_t)&ldsB[srow * 64 + scol];
  const _Float16* gbase = Bt + (size_t)(n0 + srow) * HH + scol;

  v8f acc[4];
#pragma unroll
  for (int t = 0; t < 4; ++t) acc[t] = zero8();

  for (int kc = 0; kc < HH; kc += 64) {
    // ---- stage B[n0..n0+63][kc..kc+63] -> LDS (ASYNCcnt-tracked) ----------
    const _Float16* gsrc = gbase + kc;
    asm volatile(
        "global_load_async_to_lds_b128 %0, %1, off\n\t"
        "global_load_async_to_lds_b128 %0, %1, off offset:16"
        :
        : "v"(ldsaddr), "v"(gsrc)
        : "memory");
    asm volatile("s_wait_asynccnt 0x0" ::: "memory");
    __syncthreads();

    if (kc + 128 < HH)
      __builtin_prefetch((const void*)(Arow + kc + 128), 0, 0);

#pragma unroll
    for (int kk = 0; kk < 64; kk += 32) {
      v16h a = loadA(Arow + kc + kk, half);
#pragma unroll
      for (int t = 0; t < 4; ++t) {
        const _Float16* bp = &ldsB[(t * 16 + lm) * 64 + kk];
        acc[t] = wmma_f16(a, loadB(bp, half), acc[t]);
      }
    }
    __syncthreads();   // before overwriting LDS chunk
  }

#pragma unroll
  for (int t = 0; t < 4; ++t) {
    const int n = n0 + t * 16 + lm;
    const float bv = bias[n];
#pragma unroll
    for (int r = 0; r < 8; ++r) {
      const int m = m0 + r + half * 8;
      float v = acc[t][r] + bv;
      if (MODE == 0) {                 // [b][h][s][d] f16
        int b = m / SS, s = m - b * SS, h = n >> 6, d = n & 63;
        ((_Float16*)outp)[(((size_t)b * NH + h) * SS + s) * DD + d] = (_Float16)v;
      } else if (MODE == 1) {          // [b][h][d][s] f16 (V^T)
        int b = m / SS, s = m - b * SS, h = n >> 6, d = n & 63;
        ((_Float16*)outp)[(((size_t)b * NH + h) * DD + d) * SS + s] = (_Float16)v;
      } else if (MODE == 2) {          // gelu, f16 row-major
        float g = 0.5f * v * (1.0f + erff(v * 0.70710678118654752f));
        ((_Float16*)outp)[(size_t)m * HH + n] = (_Float16)g;
      } else {                         // f32 row-major
        ((float*)outp)[(size_t)m * HH + n] = v;
      }
    }
  }
}

// ---------------------------------------------------------------------------
// Attention: one wave per (b, head, 16-query tile). scores->softmax->P*V,
// all matrix math through WMMA. 2 waves per block, 48KB LDS.
// ---------------------------------------------------------------------------
__global__ __launch_bounds__(64) void attn_kernel(const _Float16* __restrict__ qg,
                                                  const _Float16* __restrict__ kg,
                                                  const _Float16* __restrict__ vT,
                                                  const int* __restrict__ lengths,
                                                  float* __restrict__ attout) {
  __shared__ float    sc[2][16][256];
  __shared__ _Float16 pr[2][16][256];

  const int w = threadIdx.x >> 5;
  const int lane = threadIdx.x & 31;
  const int half = lane >> 4;
  const int lm = lane & 15;

  const int gw = blockIdx.x * 2 + w;           // 0..7679
  const int qt = gw % 15;                      // query tile (16 rows), S=240
  const int t2 = gw / 15;
  const int h = t2 & 15;
  const int b = t2 >> 4;

  const _Float16* qh = qg + (size_t)(b * NH + h) * SS * DD;
  const _Float16* kh = kg + (size_t)(b * NH + h) * SS * DD;
  const _Float16* vh = vT + (size_t)(b * NH + h) * DD * SS;

  // ---- scores = q @ k^T * (1/sqrt(H)), causal tiles kt <= qt --------------
  const _Float16* qrowp = qh + (size_t)(qt * 16 + lm) * DD;
  v16h a0 = loadA(qrowp, half);
  v16h a1 = loadA(qrowp + 32, half);

  for (int kt = 0; kt <= qt; ++kt) {
    const _Float16* kbp = kh + (size_t)(kt * 16 + lm) * DD;  // B col = key index
    v8f acc = zero8();
    acc = wmma_f16(a0, loadB(kbp, half), acc);
    acc = wmma_f16(a1, loadB(kbp + 32, half), acc);
    const int kkcol = kt * 16 + lm;
#pragma unroll
    for (int r = 0; r < 8; ++r) {
      const int qrow = qt * 16 + r + half * 8;
      float sv = acc[r] * 0.03125f;              // 1/sqrt(1024)
      if (kkcol > qrow) sv = -__builtin_inff();  // causal
      sc[w][r + half * 8][kkcol] = sv;
    }
  }
  __syncthreads();

  // ---- softmax per row (2 lanes per row), zero padded query rows ----------
  const int row = lane & 15;
  const int seg = lane >> 4;
  const int nk = (qt + 1) * 16;
  const int nk32 = (nk + 31) & ~31;
  const int qglob = qt * 16 + row;

  float mx = -__builtin_inff();
  for (int c = seg; c < nk; c += 2) mx = fmaxf(mx, sc[w][row][c]);
  mx = fmaxf(mx, __shfl_xor(mx, 16));
  float sum = 0.0f;
  for (int c = seg; c < nk; c += 2) sum += __expf(sc[w][row][c] - mx);
  sum += __shfl_xor(sum, 16);
  const float validf = (qglob < lengths[b]) ? 1.0f : 0.0f;
  const float inv = validf / sum;                // diag always valid -> sum>=1
  for (int c = seg; c < nk; c += 2)
    pr[w][row][c] = (_Float16)(__expf(sc[w][row][c] - mx) * inv);
  for (int c = nk + seg; c < nk32; c += 2)
    pr[w][row][c] = (_Float16)0.0f;
  __syncthreads();

  // ---- out = P @ V  (A from LDS probs, B from V^T) ------------------------
  v8f acc2[4];
#pragma unroll
  for (int t = 0; t < 4; ++t) acc2[t] = zero8();

  const _Float16* prow = &pr[w][lm][0];
  for (int kk0 = 0; kk0 < nk32; kk0 += 32) {
    v16h a = loadA(prow + kk0, half);
#pragma unroll
    for (int t = 0; t < 4; ++t) {
      const _Float16* vb = vh + (size_t)(t * 16 + lm) * SS + kk0;
      acc2[t] = wmma_f16(a, loadB(vb, half), acc2[t]);
    }
  }

#pragma unroll
  for (int t = 0; t < 4; ++t) {
    const int col = h * DD + t * 16 + lm;
#pragma unroll
    for (int r = 0; r < 8; ++r) {
      const int s = qt * 16 + r + half * 8;
      attout[((size_t)b * SS + s) * HH + col] = acc2[t][r];
    }
  }
}

// ---------------------------------------------------------------------------
// Residual + LayerNorm: y = LN(a + b) * gamma + beta ; emits f32 + f16 copies
// ---------------------------------------------------------------------------
__global__ __launch_bounds__(256) void ln_kernel(const float* __restrict__ a,
                                                 const float* __restrict__ bsrc,
                                                 const float* __restrict__ gamma,
                                                 const float* __restrict__ beta,
                                                 float* __restrict__ y,
                                                 _Float16* __restrict__ yh) {
  const int row = blockIdx.x;
  const int t = threadIdx.x;
  __shared__ float red[256], red2[256];
  const float* pa = a + (size_t)row * HH;
  const float* pb = bsrc + (size_t)row * HH;
  float s = 0.0f, s2 = 0.0f;
  float xv[4];
#pragma unroll
  for (int i = 0; i < 4; ++i) {
    int h = t * 4 + i;
    float x = pa[h] + pb[h];
    xv[i] = x;
    s += x;
    s2 += x * x;
  }
  red[t] = s; red2[t] = s2;
  __syncthreads();
  for (int off = 128; off; off >>= 1) {
    if (t < off) { red[t] += red[t + off]; red2[t] += red2[t + off]; }
    __syncthreads();
  }
  const float mean = red[0] * (1.0f / HH);
  const float var = red2[0] * (1.0f / HH) - mean * mean;
  const float rstd = rsqrtf(var + 1e-5f);
#pragma unroll
  for (int i = 0; i < 4; ++i) {
    int h = t * 4 + i;
    float o = (xv[i] - mean) * rstd * gamma[h] + beta[h];
    y[(size_t)row * HH + h] = o;
    yh[(size_t)row * HH + h] = (_Float16)o;
  }
}

// ---------------------------------------------------------------------------
// Mean-pool over S (padding included, faithful) -> FC(1024x4) -> softmax
// ---------------------------------------------------------------------------
__global__ __launch_bounds__(256) void pool_kernel(const float* __restrict__ z,
                                                   const float* __restrict__ Wfc,
                                                   const float* __restrict__ bfc,
                                                   float* __restrict__ out) {
  const int b = blockIdx.x;
  const int t = threadIdx.x;
  float acc[4] = {0.f, 0.f, 0.f, 0.f};
#pragma unroll
  for (int i = 0; i < 4; ++i) {
    int c = t * 4 + i;
    const float* zp = z + (size_t)b * SS * HH + c;
    float s = 0.0f;
    for (int si = 0; si < SS; ++si) s += zp[(size_t)si * HH];
    s *= (1.0f / SS);
#pragma unroll
    for (int j = 0; j < 4; ++j) acc[j] += s * Wfc[c * 4 + j];
  }
  __shared__ float red[256][4];
#pragma unroll
  for (int j = 0; j < 4; ++j) red[t][j] = acc[j];
  __syncthreads();
  for (int off = 128; off; off >>= 1) {
    if (t < off) {
#pragma unroll
      for (int j = 0; j < 4; ++j) red[t][j] += red[t + off][j];
    }
    __syncthreads();
  }
  if (t == 0) {
    float lg[4], mx = -__builtin_inff();
#pragma unroll
    for (int j = 0; j < 4; ++j) { lg[j] = red[0][j] + bfc[j]; mx = fmaxf(mx, lg[j]); }
    float s = 0.0f;
#pragma unroll
    for (int j = 0; j < 4; ++j) { lg[j] = __expf(lg[j] - mx); s += lg[j]; }
#pragma unroll
    for (int j = 0; j < 4; ++j) out[b * 4 + j] = lg[j] / s;
  }
}

// ---------------------------------------------------------------------------
// Host orchestration
// ---------------------------------------------------------------------------
extern "C" void kernel_launch(void* const* d_in, const int* in_sizes, int n_in,
                              void* d_out, int out_size, void* d_ws, size_t ws_size,
                              hipStream_t stream) {
  (void)in_sizes; (void)n_in; (void)out_size; (void)ws_size;

  const float* x    = (const float*)d_in[0];
  const float* wemb = (const float*)d_in[1];
  const float* pemb = (const float*)d_in[2];
  const float* Wq   = (const float*)d_in[3];
  const float* bq   = (const float*)d_in[4];
  const float* Wk   = (const float*)d_in[5];
  const float* bk   = (const float*)d_in[6];
  const float* Wv   = (const float*)d_in[7];
  const float* bv   = (const float*)d_in[8];
  const float* W1   = (const float*)d_in[9];
  const float* b1   = (const float*)d_in[10];
  const float* W2   = (const float*)d_in[11];
  const float* b2   = (const float*)d_in[12];
  const float* ln1s = (const float*)d_in[13];
  const float* ln1b = (const float*)d_in[14];
  const float* ln2s = (const float*)d_in[15];
  const float* ln2b = (const float*)d_in[16];
  const float* Wfc  = (const float*)d_in[17];
  const float* bfc  = (const float*)d_in[18];
  const int* lengths = (const int*)d_in[19];
  float* out = (float*)d_out;

  unsigned char* base = (unsigned char*)d_ws;
  size_t off = 0;
  auto alloc = [&](size_t bytes) -> void* {
    off = (off + 255) & ~(size_t)255;
    void* p = base + off;
    off += bytes;
    return p;
  };

  const size_t WT = (size_t)LL * HH * HH * sizeof(_Float16);   // 16 MB each
  _Float16* WqT = (_Float16*)alloc(WT);
  _Float16* WkT = (_Float16*)alloc(WT);
  _Float16* WvT = (_Float16*)alloc(WT);
  _Float16* W1T = (_Float16*)alloc(WT);
  _Float16* W2T = (_Float16*)alloc(WT);

  float*    zf   = (float*)   alloc((size_t)MM * HH * 4);
  _Float16* zh   = (_Float16*)alloc((size_t)MM * HH * 2);
  _Float16* qb   = (_Float16*)alloc((size_t)MM * HH * 2);
  _Float16* kb   = (_Float16*)alloc((size_t)MM * HH * 2);
  _Float16* vTb  = (_Float16*)alloc((size_t)MM * HH * 2 + 256); // pad: nk32 overread
  float*    attn = (float*)   alloc((size_t)MM * HH * 4);
  float*    z1   = (float*)   alloc((size_t)MM * HH * 4);
  _Float16* z1h  = (_Float16*)alloc((size_t)MM * HH * 2);
  _Float16* ffh  = (_Float16*)alloc((size_t)MM * HH * 2);
  float*    ff2  = (float*)   alloc((size_t)MM * HH * 4);

  // Weights f32 -> f16 transposed (done every call; deterministic)
  const int tot = LL * HH * HH;
  const int cblk = tot / 256;
  convT_kernel<<<cblk, 256, 0, stream>>>(Wq, WqT, tot);
  convT_kernel<<<cblk, 256, 0, stream>>>(Wk, WkT, tot);
  convT_kernel<<<cblk, 256, 0, stream>>>(Wv, WvT, tot);
  convT_kernel<<<cblk, 256, 0, stream>>>(W1, W1T, tot);
  convT_kernel<<<cblk, 256, 0, stream>>>(W2, W2T, tot);

  embed_kernel<<<MM, 256, 0, stream>>>(x, wemb, pemb, lengths, zf, zh);

  const dim3 gg(MM / 128, HH / 64);   // 60 x 16 blocks, 8 waves each
  for (int i = 0; i < LL; ++i) {
    const size_t wo = (size_t)i * HH * HH;
    gemm_kernel<0><<<gg, 256, 0, stream>>>(zh, WqT + wo, bq + i * HH, qb);
    gemm_kernel<0><<<gg, 256, 0, stream>>>(zh, WkT + wo, bk + i * HH, kb);
    gemm_kernel<1><<<gg, 256, 0, stream>>>(zh, WvT + wo, bv + i * HH, vTb);
    attn_kernel<<<BB * NH * 15 / 2, 64, 0, stream>>>(qb, kb, vTb, lengths, attn);
    ln_kernel<<<MM, 256, 0, stream>>>(zf, attn, ln1s + i * HH, ln1b + i * HH, z1, z1h);
    gemm_kernel<2><<<gg, 256, 0, stream>>>(z1h, W1T + wo, b1 + i * HH, ffh);
    gemm_kernel<3><<<gg, 256, 0, stream>>>(ffh, W2T + wo, b2 + i * HH, ff2);
    ln_kernel<<<MM, 256, 0, stream>>>(z1, ff2, ln2s + i * HH, ln2b + i * HH, zf, zh);
  }

  pool_kernel<<<BB, 256, 0, stream>>>(zf, Wfc, bfc, out);
}
